// _Relu2MlpPair_58265526338194
// MI455X (gfx1250) — 8-bit matrix operand order repaired and checked by reading, compiled, not run on hardware
//
#include <hip/hip_runtime.h>
#include <cstdint>


typedef __attribute__((ext_vector_type(8))) int v8i;

#define TOKENS   32768
#define DDIM     1024
#define FDIM     3072
#define MTILES   (TOKENS / 16)
#define KT1      (DDIM / 64)
#define KT2      (FDIM / 64)
#define NT1      (FDIM / 16)
#define WCOUNT   (FDIM * DDIM)
#define EPSQ     1e-5f

#define WS_SUMS  0u
#define WS_PART  256u
#define WS_AS    2048u
#define WS_HS    133120u
#define WS_UPW   264192u
#define WS_DNW   3409920u
#define WS_QX    6555648u
#define WS_QH    40110080u

__device__ __forceinline__ int kmapA(int half, int v, int b) {
  return ((v & 4) << 3) | ((v & 2) << 3) | (half << 3) | ((v & 1) << 2) | b;
}
__device__ __forceinline__ int kmapB(int half, int v, int b) {
  return ((v & 4) << 3) | ((v & 2) << 3) | (half << 3) | ((v & 1) << 2) | b;
}

__global__ __launch_bounds__(256) void k_wsum_partial(
    const float* __restrict__ up_w, const float* __restrict__ down_w,
    float* __restrict__ part) {
  int which = blockIdx.x >> 7;
  int bid   = blockIdx.x & 127;
  const float* w = which ? down_w : up_w;
  float s = 0.f;
  for (int i = bid * 256 + threadIdx.x; i < WCOUNT; i += 128 * 256)
    s += fabsf(w[i]);
  __shared__ float red[256];
  red[threadIdx.x] = s;
  __syncthreads();
  for (int off = 128; off > 0; off >>= 1) {
    if (threadIdx.x < (unsigned)off) red[threadIdx.x] += red[threadIdx.x + off];
    __syncthreads();
  }
  if (threadIdx.x == 0) part[which * 128 + bid] = red[0];
}

__global__ void k_wsum_final(const float* __restrict__ part,
                             float* __restrict__ sums) {
  int t = threadIdx.x;
  if (t < 2) {
    float s = 0.f;
    for (int i = 0; i < 128; ++i) s += part[t * 128 + i];
    sums[t] = s;
  }
}

__global__ __launch_bounds__(256) void k_wpack(
    const float* __restrict__ up_w, const float* __restrict__ down_w,
    const float* __restrict__ sums,
    unsigned* __restrict__ upq, unsigned* __restrict__ dnq) {
  const int PER = NT1 * KT1 * 32 * 8;
  int dd = blockIdx.x * 256 + threadIdx.x;
  int which = (dd >= PER);
  int local = which ? dd - PER : dd;

  int kt_tiles   = which ? KT2 : KT1;
  int rowstride  = which ? FDIM : DDIM;
  const float* w = which ? down_w : up_w;
  unsigned* dst  = which ? dnq : upq;

  int per_ntile = kt_tiles * 256;
  int nt   = local / per_ntile;
  int rem  = local - nt * per_ntile;
  int kt   = rem >> 8;
  int ln   = (rem >> 3) & 31;
  int v    = rem & 7;
  int n    = nt * 16 + (ln & 15);
  int hf   = ln >> 4;

  float mean = fmaxf(sums[which] * (1.0f / (float)WCOUNT), EPSQ);
  float inv_mean = 1.0f / mean;

  unsigned pk = 0;
#pragma unroll
  for (int b = 0; b < 4; ++b) {
    int k = kt * 64 + kmapB(hf, v, b);
    float wv = w[(size_t)n * rowstride + k];
    int q = (int)rintf(wv * inv_mean);
    q = q > 1 ? 1 : (q < -1 ? -1 : q);
    pk |= ((unsigned)(q & 0xff)) << (8 * b);
  }
  dst[local] = pk;
}

__global__ __launch_bounds__(256) void k_actquant(
    const float* __restrict__ x, float* __restrict__ aS,
    unsigned* __restrict__ qx) {
  __shared__ float red[16][16];
  __shared__ float s_qs[16];
  int tid = threadIdx.x, mTile = blockIdx.x;
  int tok = tid >> 4, j = tid & 15;

  const float* xr = x + (size_t)(mTile * 16 + tok) * DDIM;
  float pm = 0.f;
  for (int k = j; k < DDIM; k += 16) pm = fmaxf(pm, fabsf(xr[k]));
  red[tok][j] = pm;
  __syncthreads();
  if (tid < 16) {
    float mx = 0.f;
    for (int t = 0; t < 16; ++t) mx = fmaxf(mx, red[tid][t]);
    float cl = fmaxf(mx, EPSQ);
    s_qs[tid] = 127.f / cl;
    aS[mTile * 16 + tid] = cl * (1.f / 127.f);
  }
  __syncthreads();

  unsigned* qo = qx + (size_t)mTile * (KT1 * 256);
  for (int dd = tid; dd < KT1 * 256; dd += 256) {
    int kt = dd >> 8, ln = (dd >> 3) & 31, v = dd & 7;
    int m = ln & 15, hf = ln >> 4;
    const float* xm = x + (size_t)(mTile * 16 + m) * DDIM + kt * 64;
    float s = s_qs[m];
    unsigned pk = 0;
#pragma unroll
    for (int b = 0; b < 4; ++b) {
      int q = (int)rintf(xm[kmapA(hf, v, b)] * s);
      q = q > 127 ? 127 : (q < -128 ? -128 : q);
      pk |= ((unsigned)(q & 0xff)) << (8 * b);
    }
    qo[dd] = pk;
  }
}

__global__ __launch_bounds__(256) void k_gemm1(
    const unsigned* __restrict__ qx, const unsigned* __restrict__ upq,
    const float* __restrict__ aS, const float* __restrict__ sums,
    const float* __restrict__ up_b,
    unsigned* __restrict__ qh, float* __restrict__ hS) {
  __shared__ _Float16 hbuf[16][FDIM];
  __shared__ float red[16][16];
  __shared__ float s_as[16];
  __shared__ float s_qs[16];

  int tid = threadIdx.x, wave = tid >> 5, lane = tid & 31;
  int mTile = blockIdx.x;
  int half = lane >> 4, ncol = lane & 15;

  float wm_up = fmaxf(sums[0] * (1.0f / (float)WCOUNT), EPSQ);
  if (tid < 16) s_as[tid] = aS[mTile * 16 + tid] * wm_up;
  __syncthreads();

  const v8i* ap = (const v8i*)qx;
  const v8i* bp = (const v8i*)upq;
  v8i a[KT1];
#pragma unroll
  for (int kt = 0; kt < KT1; ++kt) a[kt] = ap[(mTile * KT1 + kt) * 32 + lane];

  for (int i = 0; i < NT1 / 8; ++i) {
    int nt = wave * (NT1 / 8) + i;
    v8i acc = {0, 0, 0, 0, 0, 0, 0, 0};
#pragma unroll
    for (int kt = 0; kt < KT1; ++kt) {
      v8i bf = bp[(nt * KT1 + kt) * 32 + lane];
      acc = __builtin_amdgcn_wmma_i32_16x16x64_iu8(
          true, a[kt], true, bf, acc, false, false);
    }
    int n = nt * 16 + ncol;
    float bias = up_b[n];
#pragma unroll
    for (int v = 0; v < 8; ++v) {
      int m = v + 8 * half;
      float y = (float)acc[v] * s_as[m] + bias;
      float r = fmaxf(y, 0.f);
      hbuf[m][n] = (_Float16)(r * r);
    }
  }
  __syncthreads();

  int tok = tid >> 4, j = tid & 15;
  float pm = 0.f;
  for (int f = j; f < FDIM; f += 16) pm = fmaxf(pm, (float)hbuf[tok][f]);
  red[tok][j] = pm;
  __syncthreads();
  if (tid < 16) {
    float mx = 0.f;
    for (int t = 0; t < 16; ++t) mx = fmaxf(mx, red[tid][t]);
    float cl = fmaxf(mx, EPSQ);
    s_qs[tid] = 127.f / cl;
    hS[mTile * 16 + tid] = cl * (1.f / 127.f);
  }
  __syncthreads();

  unsigned* qo = qh + (size_t)mTile * (KT2 * 256);
  for (int dd = tid; dd < KT2 * 256; dd += 256) {
    int kt = dd >> 8, ln = (dd >> 3) & 31, v = dd & 7;
    int m = ln & 15, hf = ln >> 4;
    float s = s_qs[m];
    unsigned pk = 0;
#pragma unroll
    for (int b = 0; b < 4; ++b) {
      int f = kt * 64 + kmapA(hf, v, b);
      int q = (int)rintf((float)hbuf[m][f] * s);
      q = q > 127 ? 127 : (q < -128 ? -128 : q);
      pk |= ((unsigned)(q & 0xff)) << (8 * b);
    }
    qo[dd] = pk;
  }
}

__global__ __launch_bounds__(256) void k_gemm2(
    const unsigned* __restrict__ qh, const unsigned* __restrict__ dnq,
    const float* __restrict__ hS, const float* __restrict__ sums,
    const float* __restrict__ down_b, float* __restrict__ out) {
  __shared__ float s_hs[32];
  int tid = threadIdx.x, wave = tid >> 5, lane = tid & 31;
  int half = lane >> 4, ncol = lane & 15;

  int mT0 = blockIdx.x * 2;
  int ntBase = (blockIdx.y * 8 + wave) * 4;

  float wm_dn = fmaxf(sums[1] * (1.0f / (float)WCOUNT), EPSQ);
  if (tid < 32) s_hs[tid] = hS[blockIdx.x * 32 + tid] * wm_dn;
  __syncthreads();

  const v8i* ap = (const v8i*)qh;
  const v8i* bp = (const v8i*)dnq;

  v8i acc[2][4];
#pragma unroll
  for (int s = 0; s < 2; ++s)
#pragma unroll
    for (int j = 0; j < 4; ++j) acc[s][j] = (v8i){0, 0, 0, 0, 0, 0, 0, 0};

  for (int kt = 0; kt < KT2; ++kt) {
    v8i af0 = ap[((size_t)(mT0 + 0) * KT2 + kt) * 32 + lane];
    v8i af1 = ap[((size_t)(mT0 + 1) * KT2 + kt) * 32 + lane];
#pragma unroll
    for (int j = 0; j < 4; ++j) {
      v8i bf = bp[((ntBase + j) * KT2 + kt) * 32 + lane];
      acc[0][j] = __builtin_amdgcn_wmma_i32_16x16x64_iu8(
          true, af0, true, bf, acc[0][j], false, false);
      acc[1][j] = __builtin_amdgcn_wmma_i32_16x16x64_iu8(
          true, af1, true, bf, acc[1][j], false, false);
    }
  }

#pragma unroll
  for (int s = 0; s < 2; ++s) {
#pragma unroll
    for (int j = 0; j < 4; ++j) {
      int n = (ntBase + j) * 16 + ncol;
      float bias = down_b[n];
#pragma unroll
      for (int v = 0; v < 8; ++v) {
        int m = v + 8 * half;
        int token = (mT0 + s) * 16 + m;
        out[(size_t)token * DDIM + n] =
            (float)acc[s][j][v] * s_hs[s * 16 + m] + bias;
      }
    }
  }
}

extern "C" void kernel_launch(void* const* d_in, const int* in_sizes, int n_in,
                              void* d_out, int out_size, void* d_ws,
                              size_t ws_size, hipStream_t stream) {
  const float* x      = (const float*)d_in[0];
  const float* up_w   = (const float*)d_in[1];
  const float* up_b   = (const float*)d_in[2];
  const float* down_w = (const float*)d_in[3];
  const float* down_b = (const float*)d_in[4];
  float* out = (float*)d_out;

  char* ws = (char*)d_ws;
  float*    sums = (float*)(ws + WS_SUMS);
  float*    part = (float*)(ws + WS_PART);
  float*    aS   = (float*)(ws + WS_AS);
  float*    hS   = (float*)(ws + WS_HS);
  unsigned* upq  = (unsigned*)(ws + WS_UPW);
  unsigned* dnq  = (unsigned*)(ws + WS_DNW);
  unsigned* qx   = (unsigned*)(ws + WS_QX);
  unsigned* qh   = (unsigned*)(ws + WS_QH);

  k_wsum_partial<<<256, 256, 0, stream>>>(up_w, down_w, part);
  k_wsum_final<<<1, 64, 0, stream>>>(part, sums);
  k_wpack<<<(2 * NT1 * KT1 * 32 * 8) / 256, 256, 0, stream>>>(up_w, down_w,
                                                              sums, upq, dnq);
  k_actquant<<<MTILES, 256, 0, stream>>>(x, aS, qx);
  k_gemm1<<<MTILES, 256, 0, stream>>>(qx, upq, aS, sums, up_b, qh, hS);
  k_gemm2<<<dim3(MTILES / 2, 2), 256, 0, stream>>>(qh, dnq, hS, sums, down_b,
                                                   out);
}
